// QuantizerModule_55989193671193
// MI455X (gfx1250) — matrix operand order repaired and checked by reading, not compiled, not run on hardware
//
#include <hip/hip_runtime.h>

#define N_E    2048
#define E_DIM  512
#define N_TOK  32768

#define MT          4                 // 16-row M-tiles per block -> 64 tokens/block
#define TOK_PER_BLK (MT * 16)
#define WAVES       8
#define THREADS     (WAVES * 32)
#define NT_PER_WAVE (N_E / (WAVES * 16))   // 16 N-tiles (256 codes) per wave
#define KSTEPS      (E_DIM / 32)

typedef __bf16 v16bf __attribute__((ext_vector_type(16)));
typedef float  v8f   __attribute__((ext_vector_type(8)));

union FragBF { uint4 q[2]; v16bf v; };

__device__ __forceinline__ unsigned f2bf(float f) {   // RNE f32 -> bf16 bits
    unsigned u = __float_as_uint(f);
    return (u + 0x7fffu + ((u >> 16) & 1u)) >> 16;
}

__device__ __forceinline__ unsigned long long shflxor64(unsigned long long x, int m) {
    unsigned lo = __shfl_xor((unsigned)x, m, 32);
    unsigned hi = __shfl_xor((unsigned)(x >> 32), m, 32);
    return ((unsigned long long)hi << 32) | lo;
}

// ---------------- pre-pass: W f32 -> bf16, plus ||w||^2 per code ----------------
__global__ __launch_bounds__(THREADS)
void conv_kernel(const float* __restrict__ W, __bf16* __restrict__ Wb,
                 float* __restrict__ wnorm) {
    const int lane = threadIdx.x & 31;
    const int wv   = threadIdx.x >> 5;
    const int row  = blockIdx.x * WAVES + wv;

    const float4* src = (const float4*)(W + (size_t)row * E_DIM);
    uint2* dst = (uint2*)(Wb + (size_t)row * E_DIM);
    float s = 0.f;
#pragma unroll
    for (int i = 0; i < E_DIM / (32 * 4); ++i) {
        float4 f = src[lane + i * 32];
        s = fmaf(f.x, f.x, fmaf(f.y, f.y, fmaf(f.z, f.z, fmaf(f.w, f.w, s))));
        uint2 p;
        p.x = f2bf(f.x) | (f2bf(f.y) << 16);
        p.y = f2bf(f.z) | (f2bf(f.w) << 16);
        dst[lane + i * 32] = p;
    }
#pragma unroll
    for (int m = 16; m; m >>= 1) s += __shfl_xor(s, m, 32);
    if (lane == 0) wnorm[row] = s;
}

// ---------------- main: WMMA distances + block-local argmin + gather ----------------
// LDS: 64KB, A stored fragment-packed: per (mt,kstep) 1KB = [chunk][lane][8 bf16]
// After the main loop the same memory is reused for the argmin scratch.
__global__ __launch_bounds__(THREADS)
void vq_argmin_kernel(const float* __restrict__ x, const float* __restrict__ W,
                      const __bf16* __restrict__ Wb, const float* __restrict__ wnorm,
                      float* __restrict__ zq, int* __restrict__ idx_out) {
    __shared__ __align__(16) unsigned char sMem[MT * 16 * E_DIM * 2];  // 65536 B
    unsigned long long* sRed = (unsigned long long*)sMem;              // 4096 B
    int* sBest = (int*)(sMem + TOK_PER_BLK * WAVES * 8);               // 256 B

    const int tid  = threadIdx.x;
    const int lane = tid & 31;
    const int wv   = tid >> 5;
    const int l16  = lane & 15;
    const int half = lane >> 4;
    const int tok0 = blockIdx.x * TOK_PER_BLK;

    // ---- stage x strip: global f32 -> LDS bf16, fragment-packed ----
    const float4* x4 = (const float4*)(x + (size_t)tok0 * E_DIM);
    for (int i = tid; i < TOK_PER_BLK * (E_DIM / 4); i += THREADS) {
        const int row = i >> 7;            // E_DIM/4 == 128
        const int c4  = i & 127;
        float4 f = x4[row * 128 + c4];
        uint2 p;
        p.x = f2bf(f.x) | (f2bf(f.y) << 16);
        p.y = f2bf(f.z) | (f2bf(f.w) << 16);
        // element position -> fragment-packed address
        const int kk     = (c4 * 4) & 31;          // K within 32-step
        const int kstep  = c4 >> 3;
        const int khalf  = (kk >> 3) & 1;
        const int chunk  = (kk >> 4) & 1;
        const int pos    = kk & 7;
        const int lane_t = (row & 15) + (khalf << 4);
        const int off = (((row >> 4) * 16 + kstep) << 10) + (chunk << 9)
                        + (lane_t << 4) + (pos << 1);
        *(uint2*)(sMem + off) = p;
    }
    __syncthreads();

    // ---- per-lane running argmin (deferred key packing) ----
    float    bestS[MT][8];
    unsigned bestI[MT][8];
#pragma unroll
    for (int mt = 0; mt < MT; ++mt)
#pragma unroll
        for (int v = 0; v < 8; ++v) { bestS[mt][v] = __builtin_inff(); bestI[mt][v] = 0u; }

    const v8f vzero = {0.f, 0.f, 0.f, 0.f, 0.f, 0.f, 0.f, 0.f};

    for (int nt = 0; nt < NT_PER_WAVE; ++nt) {
        const int n0 = (wv * NT_PER_WAVE + nt) * 16;
        v8f acc[MT];
#pragma unroll
        for (int mt = 0; mt < MT; ++mt) acc[mt] = vzero;

        const __bf16* brow = Wb + (size_t)(n0 + l16) * E_DIM + half * 8;
        const float wn = wnorm[n0 + l16];

#pragma unroll 4
        for (int k0 = 0; k0 < E_DIM; k0 += 32) {
            FragBF B;
            B.q[0] = *(const uint4*)(brow + k0);
            B.q[1] = *(const uint4*)(brow + k0 + 16);
            const int kbase = (k0 << 5) + (lane << 4);   // kstep*1024 + lane*16
#pragma unroll
            for (int mt = 0; mt < MT; ++mt) {
                FragBF A;
                const unsigned char* ap = sMem + (mt << 14) + kbase;
                A.q[0] = *(const uint4*)ap;          // chunk0: K = half*8 .. +7
                A.q[1] = *(const uint4*)(ap + 512);  // chunk1: K = 16+half*8 .. +7
                acc[mt] = __builtin_amdgcn_wmma_f32_16x16x32_bf16(
                    false, A.v, false, B.v, (short)0, acc[mt], false, false);
            }
        }

        const unsigned myIdx = (unsigned)(n0 + l16);
#pragma unroll
        for (int mt = 0; mt < MT; ++mt)
#pragma unroll
            for (int v = 0; v < 8; ++v) {
                const float s = fmaf(-2.0f, acc[mt][v], wn);
                if (s < bestS[mt][v]) { bestS[mt][v] = s; bestI[mt][v] = myIdx; }
            }
    }

    __syncthreads();   // everyone done reading sA before we alias it

    // ---- pack (ordered score | idx) once, reduce across the 16 N-lanes ----
#pragma unroll
    for (int mt = 0; mt < MT; ++mt)
#pragma unroll
        for (int v = 0; v < 8; ++v) {
            const unsigned b = __float_as_uint(bestS[mt][v]);
            const unsigned key = (b & 0x80000000u) ? ~b : (b | 0x80000000u);
            unsigned long long pk = ((unsigned long long)key << 32) | bestI[mt][v];
#pragma unroll
            for (int m = 8; m; m >>= 1) {
                unsigned long long o = shflxor64(pk, m);
                if (o < pk) pk = o;
            }
            if (l16 == 0)
                sRed[(mt * 16 + half * 8 + v) * WAVES + wv] = pk;
        }
    __syncthreads();

    // ---- cross-wave reduce: one thread per token row ----
    if (tid < TOK_PER_BLK) {
        unsigned long long b = ~0ull;
#pragma unroll
        for (int w = 0; w < WAVES; ++w) {
            unsigned long long o = sRed[tid * WAVES + w];
            if (o < b) b = o;
        }
        const int bi = (int)(b & 0xffffffffu);
        idx_out[tok0 + tid] = bi;
        sBest[tid] = bi;
    }
    __syncthreads();

    // ---- gather z_q rows from original f32 codebook ----
    float4* zq4 = (float4*)(zq + (size_t)tok0 * E_DIM);
    const float4* W4 = (const float4*)W;
    for (int i = tid; i < TOK_PER_BLK * (E_DIM / 4); i += THREADS) {
        const int row = i >> 7;
        const int c4  = i & 127;
        zq4[row * 128 + c4] = W4[(size_t)sBest[row] * 128 + c4];
    }
}

extern "C" void kernel_launch(void* const* d_in, const int* /*in_sizes*/, int /*n_in*/,
                              void* d_out, int /*out_size*/, void* d_ws,
                              size_t /*ws_size*/, hipStream_t stream) {
    const float* x = (const float*)d_in[0];
    const float* W = (const float*)d_in[1];
    float* zq = (float*)d_out;
    int* idx = (int*)(zq + (size_t)N_TOK * E_DIM);

    // workspace layout: [0, 2MB) bf16 codebook, [2MB, 2MB+8KB) wnorm f32
    __bf16* Wb   = (__bf16*)d_ws;
    float* wnorm = (float*)((char*)d_ws + (size_t)N_E * E_DIM * sizeof(__bf16));

    conv_kernel<<<N_E / WAVES, THREADS, 0, stream>>>(W, Wb, wnorm);
    vq_argmin_kernel<<<N_TOK / TOK_PER_BLK, THREADS, 0, stream>>>(x, W, Wb, wnorm,
                                                                  zq, idx);
}
